// SCNNAutoencoderModel_35785667510309
// MI455X (gfx1250) — matrix operand order repaired and checked by reading, not compiled, not run on hardware
//
#include <hip/hip_runtime.h>
#include <cstddef>
#include <cstdint>

// ---------------------------------------------------------------------------
// CDNA5 / gfx1250 types
// ---------------------------------------------------------------------------
typedef __attribute__((ext_vector_type(16))) __bf16 v16bf;
typedef __attribute__((ext_vector_type(8)))  float  v8f;
typedef unsigned short u16;

#define THR      1.0f
#define DECAY    0.99f
#define POOL_THR 0.75f
#define STEPS    10
#define BF16_ONE ((u16)0x3F80)

__device__ __forceinline__ u16 f2bf(float f) {
  union { float f; unsigned u; } x; x.f = f;
  unsigned r = x.u + 0x7FFFu + ((x.u >> 16) & 1u);   // round-to-nearest-even
  return (u16)(r >> 16);
}
__device__ __forceinline__ float bf2f(u16 b) {
  union { unsigned u; float f; } x; x.u = ((unsigned)b) << 16;
  return x.f;
}

union BFrag { v16bf v; u16 u[16]; };

// ---------------------------------------------------------------------------
// Weight convert + pad (fp32 -> bf16, zero-padded to [dstRows, dstK])
// ---------------------------------------------------------------------------
__global__ void cvt_pad_kernel(const float* __restrict__ src, u16* __restrict__ dst,
                               int srcRows, int srcK, int dstRows, int dstK) {
  int i = blockIdx.x * blockDim.x + threadIdx.x;
  int total = dstRows * dstK;
  if (i >= total) return;
  int r = i / dstK, k = i - r * dstK;
  float v = (r < srcRows && k < srcK) ? src[r * srcK + k] : 0.0f;
  dst[i] = f2bf(v);
}

// ---------------------------------------------------------------------------
// Implicit-GEMM 5x5 SAME conv (+ optional folded 2x upsample on input) with
// fused LIF epilogue.  One wave computes one 16x16 C tile via
// v_wmma_f32_16x16x32_bf16.
//   A: M = Bn*H*W output pixels, K = Cin*25 (padded to Kpad, mult of 32)
//   B: weights [N, Kpad] bf16 row-major (N mult of 16)
// mode 0: mem += cur; LIF; write bf16 spike (NCHW)
// mode 1: lcm = 0.99*(lcm + cur) for n < CoutReal   (final conv)
// ---------------------------------------------------------------------------
__global__ void __launch_bounds__(256) conv_wmma_lif(
    const u16* __restrict__ src, const u16* __restrict__ Wb,
    const float* __restrict__ bias, float* __restrict__ mem,
    u16* __restrict__ spk, float* __restrict__ lcm,
    int Bn, int Cin, int H, int W, int srcH, int srcW, int us,
    int N, int Kpad, int CoutReal, int mode)
{
  const int HW = H * W;
  const int Mt = (Bn * HW) >> 4;
  const int Nt = N >> 4;
  int wid = blockIdx.x * 8 + (threadIdx.x >> 5);
  if (wid >= Mt * Nt) return;                 // wave-uniform exit (EXEC stays full)
  int ntile = wid % Nt, mtile = wid / Nt;
  int lane = threadIdx.x & 31;
  int l16  = lane & 15;
  int hi8  = (lane & 16) ? 8 : 0;             // unused; A gather also splits K by bk16
  int bk16 = (lane & 16) ? 16 : 0;            // B fragment K-offset split

  // A-row (m) decode for this lane's gather
  int m    = mtile * 16 + l16;
  int nimg = m / HW; int rem = m - nimg * HW;
  int oy   = rem / W, ox = rem - oy * W;

  const u16* bp_base = Wb + (size_t)(ntile * 16 + l16) * Kpad + bk16;

  v8f acc = {0.f,0.f,0.f,0.f,0.f,0.f,0.f,0.f};
  for (int k0 = 0; k0 < Kpad; k0 += 32) {
    BFrag a, b;
    const u16* bp = bp_base + k0;
    __builtin_prefetch(bp + 32, 0, 1);                  // next weight K-tile
#pragma unroll
    for (int j = 0; j < 16; ++j) b.u[j] = bp[j];        // contiguous 32B
#pragma unroll
    for (int j = 0; j < 16; ++j) {                      // implicit im2col gather
      int k = k0 + j + bk16;                            // contiguous k = 16h + j (matches B)
      int c = k / 25;
      u16 av = 0;
      if (c < Cin) {
        int t  = k - c * 25;
        int ky = t / 5, kx = t - ky * 5;
        int iy = oy + ky - 2, ix = ox + kx - 2;         // SAME pad
        if (iy >= 0 && iy < H && ix >= 0 && ix < W)
          av = src[(((size_t)nimg * Cin + c) * srcH + (iy >> us)) * srcW + (ix >> us)];
      }
      a.u[j] = av;
    }
    acc = __builtin_amdgcn_wmma_f32_16x16x32_bf16(false, a.v, false, b.v,
                                                  (short)0, acc, false, false);
  }

  // Epilogue: C/D layout => n = lane%16, m = r + 8*(lane>=16)
  int n   = ntile * 16 + l16;
  int mhi = (lane & 16) ? 8 : 0;
#pragma unroll
  for (int r = 0; r < 8; ++r) {
    int mm = mtile * 16 + r + mhi;
    int ni = mm / HW; int rr = mm - ni * HW;
    int yy = rr / W, xx = rr - yy * W;
    if (mode == 0) {
      float cur = acc[r] + bias[n];
      size_t idx = (((size_t)ni * N + n) * H + yy) * W + xx;
      float mv = mem[idx] + cur;
      float ex = mv > THR ? mv : 0.f;
      mem[idx] = (mv - ex) * DECAY;
      spk[idx] = (ex > 0.f) ? BF16_ONE : (u16)0;
    } else if (n < CoutReal) {
      float cur = acc[r] + bias[n];
      size_t idx = (((size_t)ni * CoutReal + n) * H + yy) * W + xx;
      lcm[idx] = DECAY * (lcm[idx] + cur);
    }
  }
}

// ---------------------------------------------------------------------------
// Dense bf16 GEMM (A[M,K] x W[N,K]^T), split-K, partials via atomicAdd.
// Inner loop unrolled x2 (all slice spans are multiples of 64 K-elements).
// ---------------------------------------------------------------------------
__global__ void __launch_bounds__(256) fc_wmma_splitk(
    const u16* __restrict__ A, const u16* __restrict__ Wb,
    float* __restrict__ acc_out, int M, int N, int K, int S, int kTilesPerSlice)
{
  int Mt = M >> 4, Nt = N >> 4;
  int total = Mt * Nt * S;
  int wid = blockIdx.x * 8 + (threadIdx.x >> 5);
  if (wid >= total) return;
  int slice = wid % S; int t = wid / S;
  int ntile = t % Nt, mtile = t / Nt;
  int lane = threadIdx.x & 31;
  int l16  = lane & 15;
  int hi8  = (lane & 16) ? 8 : 0;
  int bk16 = (lane & 16) ? 16 : 0;

  int k_begin = slice * kTilesPerSlice * 32;
  int k_end   = k_begin + kTilesPerSlice * 32;
  if (k_end > K) k_end = K;

  const u16* ap_base = A  + (size_t)(mtile * 16 + l16) * K + bk16;
  const u16* bp_base = Wb + (size_t)(ntile * 16 + l16) * K + bk16;

  v8f acc = {0.f,0.f,0.f,0.f,0.f,0.f,0.f,0.f};
  for (int k0 = k_begin; k0 < k_end; k0 += 64) {
    BFrag a0, b0, a1, b1;
    const u16* ap = ap_base + k0;
    const u16* bp = bp_base + k0;
    __builtin_prefetch(bp + 64, 0, 1);                  // next weight K-tile pair
#pragma unroll
    for (int j = 0; j < 16; ++j) {
      a0.u[j] = ap[j];      b0.u[j] = bp[j];
      a1.u[j] = ap[32 + j]; b1.u[j] = bp[32 + j];
    }
    acc = __builtin_amdgcn_wmma_f32_16x16x32_bf16(false, a0.v, false, b0.v,
                                                  (short)0, acc, false, false);
    acc = __builtin_amdgcn_wmma_f32_16x16x32_bf16(false, a1.v, false, b1.v,
                                                  (short)0, acc, false, false);
  }
  int n   = ntile * 16 + l16;
  int mhi = (lane & 16) ? 8 : 0;
#pragma unroll
  for (int r = 0; r < 8; ++r) {
    int mm = mtile * 16 + r + mhi;
    atomicAdd(&acc_out[(size_t)mm * N + n], acc[r]);
  }
}

// Fused bias + membrane + LIF for FC layers (consumes split-K partials)
__global__ void lif_fc_kernel(const float* __restrict__ acc, const float* __restrict__ bias,
                              float* __restrict__ mem, u16* __restrict__ spk,
                              int total, int N) {
  int i = blockIdx.x * blockDim.x + threadIdx.x;
  if (i >= total) return;
  int n = i % N;
  float mv = mem[i] + acc[i] + bias[n];
  float ex = mv > THR ? mv : 0.f;
  mem[i] = (mv - ex) * DECAY;
  spk[i] = (ex > 0.f) ? BF16_ONE : (u16)0;
}

// 2x2 avg-pool membrane + fire (no decay)
__global__ void pool_fire_kernel(const u16* __restrict__ in, float* __restrict__ mem,
                                 u16* __restrict__ out, int total, int Ho, int Wo) {
  int i = blockIdx.x * blockDim.x + threadIdx.x;
  if (i >= total) return;
  int x = i % Wo; int t = i / Wo; int y = t % Ho; t /= Ho;  // t = n*C + c
  int Wi = Wo * 2;
  size_t base = (((size_t)t * (Ho * 2)) + y * 2) * Wi + x * 2;
  float s = bf2f(in[base]) + bf2f(in[base + 1]) + bf2f(in[base + Wi]) + bf2f(in[base + Wi + 1]);
  float mv = mem[i] + 0.25f * s;
  float ex = mv > POOL_THR ? mv : 0.f;
  mem[i] = mv - ex;
  out[i] = (ex > 0.f) ? BF16_ONE : (u16)0;
}

__global__ void finalize_kernel(const float* __restrict__ lcm, float* __restrict__ out, int n) {
  int i = blockIdx.x * blockDim.x + threadIdx.x;
  if (i < n) out[i] = lcm[i] * (1.0f / (float)STEPS);
}

// ---------------------------------------------------------------------------
// Workspace layout (bytes, all 256-aligned). Membranes first => one memset.
// ---------------------------------------------------------------------------
#define SZ_MC1 (16ull*32*64*64*4)
#define SZ_MP1 (16ull*32*32*32*4)
#define SZ_MC2 (16ull*64*32*32*4)
#define SZ_MP2 (16ull*64*16*16*4)
#define SZ_MF1 (16ull*1024*4)
#define SZ_MF2 (16ull*256*4)
#define SZ_MD1 (16ull*1024*4)
#define SZ_MD2 (16ull*16384*4)
#define SZ_MG1 (16ull*32*32*32*4)
#define SZ_LCM (16ull*3*64*64*4)
#define OFF_MC1 0ull
#define OFF_MP1 (OFF_MC1+SZ_MC1)
#define OFF_MC2 (OFF_MP1+SZ_MP1)
#define OFF_MP2 (OFF_MC2+SZ_MC2)
#define OFF_MF1 (OFF_MP2+SZ_MP2)
#define OFF_MF2 (OFF_MF1+SZ_MF1)
#define OFF_MD1 (OFF_MF2+SZ_MF2)
#define OFF_MD2 (OFF_MD1+SZ_MD1)
#define OFF_MG1 (OFF_MD2+SZ_MD2)
#define OFF_LCM (OFF_MG1+SZ_MG1)
#define MEMB_BYTES (OFF_LCM+SZ_LCM)

#define OFF_XB   (MEMB_BYTES)
#define OFF_WC1B (OFF_XB   + 16ull*3*64*64*2)
#define OFF_WC2B (OFF_WC1B + 32ull*96*2)
#define OFF_WF1B (OFF_WC2B + 64ull*800*2)
#define OFF_WF2B (OFF_WF1B + 1024ull*16384*2)
#define OFF_WD1B (OFF_WF2B + 256ull*1024*2)
#define OFF_WD2B (OFF_WD1B + 1024ull*256*2)
#define OFF_WG1B (OFF_WD2B + 16384ull*1024*2)
#define OFF_WG2B (OFF_WG1B + 32ull*1600*2)
#define OFF_SC1  (OFF_WG2B + 16ull*800*2)
#define OFF_SP1  (OFF_SC1  + 16ull*32*64*64*2)
#define OFF_SC2  (OFF_SP1  + 16ull*32*32*32*2)
#define OFF_SP2  (OFF_SC2  + 16ull*64*32*32*2)
#define OFF_SF1  (OFF_SP2  + 16ull*64*16*16*2)
#define OFF_SF2  (OFF_SF1  + 16ull*1024*2)
#define OFF_SD1  (OFF_SF2  + 16ull*256*2)
#define OFF_SD2  (OFF_SD1  + 16ull*1024*2)
#define OFF_SG1  (OFF_SD2  + 16ull*16384*2)
#define OFF_ACC  (OFF_SG1  + 16ull*32*32*32*2)

static inline int cdiv_i(long long a, long long b) { return (int)((a + b - 1) / b); }

extern "C" void kernel_launch(void* const* d_in, const int* in_sizes, int n_in,
                              void* d_out, int out_size, void* d_ws, size_t ws_size,
                              hipStream_t stream) {
  (void)in_sizes; (void)n_in; (void)out_size; (void)ws_size;

  const float* x   = (const float*)d_in[0];
  const float* Wc1 = (const float*)d_in[1];  const float* bc1 = (const float*)d_in[2];
  const float* Wc2 = (const float*)d_in[3];  const float* bc2 = (const float*)d_in[4];
  const float* Wf1 = (const float*)d_in[5];  const float* bf1 = (const float*)d_in[6];
  const float* Wf2 = (const float*)d_in[7];  const float* bf2 = (const float*)d_in[8];
  const float* Wd1 = (const float*)d_in[9];  const float* bd1 = (const float*)d_in[10];
  const float* Wd2 = (const float*)d_in[11]; const float* bd2 = (const float*)d_in[12];
  const float* Wg1 = (const float*)d_in[13]; const float* bg1 = (const float*)d_in[14];
  const float* Wg2 = (const float*)d_in[15]; const float* bg2 = (const float*)d_in[16];
  // d_in[17] = steps (known constant 10; must be host-side for launch counts)

  char* ws = (char*)d_ws;
  float* mC1 = (float*)(ws+OFF_MC1); float* mP1 = (float*)(ws+OFF_MP1);
  float* mC2 = (float*)(ws+OFF_MC2); float* mP2 = (float*)(ws+OFF_MP2);
  float* mF1 = (float*)(ws+OFF_MF1); float* mF2 = (float*)(ws+OFF_MF2);
  float* mD1 = (float*)(ws+OFF_MD1); float* mD2 = (float*)(ws+OFF_MD2);
  float* mG1 = (float*)(ws+OFF_MG1); float* lcm = (float*)(ws+OFF_LCM);
  u16* xb   = (u16*)(ws+OFF_XB);
  u16* Wc1b = (u16*)(ws+OFF_WC1B); u16* Wc2b = (u16*)(ws+OFF_WC2B);
  u16* Wf1b = (u16*)(ws+OFF_WF1B); u16* Wf2b = (u16*)(ws+OFF_WF2B);
  u16* Wd1b = (u16*)(ws+OFF_WD1B); u16* Wd2b = (u16*)(ws+OFF_WD2B);
  u16* Wg1b = (u16*)(ws+OFF_WG1B); u16* Wg2b = (u16*)(ws+OFF_WG2B);
  u16* sc1 = (u16*)(ws+OFF_SC1); u16* sp1 = (u16*)(ws+OFF_SP1);
  u16* sc2 = (u16*)(ws+OFF_SC2); u16* sp2 = (u16*)(ws+OFF_SP2);
  u16* sf1 = (u16*)(ws+OFF_SF1); u16* sf2 = (u16*)(ws+OFF_SF2);
  u16* sd1 = (u16*)(ws+OFF_SD1); u16* sd2 = (u16*)(ws+OFF_SD2);
  u16* sg1 = (u16*)(ws+OFF_SG1); float* accb = (float*)(ws+OFF_ACC);

  // Zero all membranes (one contiguous region), capture-safe.
  hipMemsetAsync(ws + OFF_MC1, 0, MEMB_BYTES, stream);

  // Convert input + weights to bf16 (padded where needed), once per launch.
#define CVT(src, dst, sr, sk, dr, dk) \
  cvt_pad_kernel<<<cdiv_i((long long)(dr)*(dk), 256), 256, 0, stream>>>(src, dst, sr, sk, dr, dk)
  CVT(x,   xb,   1,     196608, 1,     196608);
  CVT(Wc1, Wc1b, 32,    75,     32,    96);      // K: 75 -> 96
  CVT(Wc2, Wc2b, 64,    800,    64,    800);
  CVT(Wf1, Wf1b, 1024,  16384,  1024,  16384);
  CVT(Wf2, Wf2b, 256,   1024,   256,   1024);
  CVT(Wd1, Wd1b, 1024,  256,    1024,  256);
  CVT(Wd2, Wd2b, 16384, 1024,   16384, 1024);
  CVT(Wg1, Wg1b, 32,    1600,   32,    1600);
  CVT(Wg2, Wg2b, 3,     800,    16,    800);     // N: 3 -> 16
#undef CVT

  for (int s = 0; s < STEPS; ++s) {
    // conv1: 3->32 @64x64 (input = raw image, bf16)
    conv_wmma_lif<<<1024, 256, 0, stream>>>(xb, Wc1b, bc1, mC1, sc1, nullptr,
        16, 3, 64, 64, 64, 64, 0, 32, 96, 32, 0);
    pool_fire_kernel<<<2048, 256, 0, stream>>>(sc1, mP1, sp1, 16*32*32*32, 32, 32);
    // conv2: 32->64 @32x32
    conv_wmma_lif<<<512, 256, 0, stream>>>(sp1, Wc2b, bc2, mC2, sc2, nullptr,
        16, 32, 32, 32, 32, 32, 0, 64, 800, 64, 0);
    pool_fire_kernel<<<1024, 256, 0, stream>>>(sc2, mP2, sp2, 16*64*16*16, 16, 16);
    // fc1: 16384 -> 1024  (split-K 32)
    hipMemsetAsync(accb, 0, 16ull*1024*4, stream);
    fc_wmma_splitk<<<256, 256, 0, stream>>>(sp2, Wf1b, accb, 16, 1024, 16384, 32, 16);
    lif_fc_kernel<<<64, 256, 0, stream>>>(accb, bf1, mF1, sf1, 16*1024, 1024);
    // fc2: 1024 -> 256
    hipMemsetAsync(accb, 0, 16ull*256*4, stream);
    fc_wmma_splitk<<<8, 256, 0, stream>>>(sf1, Wf2b, accb, 16, 256, 1024, 4, 8);
    lif_fc_kernel<<<16, 256, 0, stream>>>(accb, bf2, mF2, sf2, 16*256, 256);
    // d1: 256 -> 1024
    hipMemsetAsync(accb, 0, 16ull*1024*4, stream);
    fc_wmma_splitk<<<16, 256, 0, stream>>>(sf2, Wd1b, accb, 16, 1024, 256, 2, 4);
    lif_fc_kernel<<<64, 256, 0, stream>>>(accb, bd1, mD1, sd1, 16*1024, 1024);
    // d2: 1024 -> 16384
    hipMemsetAsync(accb, 0, 16ull*16384*4, stream);
    fc_wmma_splitk<<<256, 256, 0, stream>>>(sd1, Wd2b, accb, 16, 16384, 1024, 2, 16);
    lif_fc_kernel<<<1024, 256, 0, stream>>>(accb, bd2, mD2, sd2, 16*16384, 16384);
    // g1: upsample(sd2 as [16,64,16,16]) -> conv 64->32 @32x32 (upsample folded)
    conv_wmma_lif<<<256, 256, 0, stream>>>(sd2, Wg1b, bg1, mG1, sg1, nullptr,
        16, 64, 32, 32, 16, 16, 1, 32, 1600, 32, 0);
    // g2: upsample(sg1) -> conv 32->3 @64x64; accumulate lcm = 0.99*(lcm + cur)
    conv_wmma_lif<<<512, 256, 0, stream>>>(sg1, Wg2b, bg2, nullptr, nullptr, lcm,
        16, 32, 64, 64, 32, 32, 1, 16, 800, 3, 1);
  }

  finalize_kernel<<<768, 256, 0, stream>>>(lcm, (float*)d_out, 16*3*64*64);
}
